// LMCL_28827820491475
// MI455X (gfx1250) — matrix operand order repaired and checked by reading, not compiled, not run on hardware
//
#include <hip/hip_runtime.h>

typedef _Float16 v16h __attribute__((ext_vector_type(16)));
typedef _Float16 v4h  __attribute__((ext_vector_type(4)));
typedef float    v8f  __attribute__((ext_vector_type(8)));

#define S_SCALE 64.0f
#define M_MARGIN 1.35f
#define BC 64          // W rows (classes) per block
#define WPAD 8         // halves of padding per LDS row -> 16B, kills bank conflicts

union Frag16 { v16h h; float4 f[2]; };

// ---------------------------------------------------------------------------
// Kernel 1: xn = x / ||x|| as f16, and zero per-row denominator accumulators.
// grid = N blocks, 256 threads, assumes D == 1024.
// ---------------------------------------------------------------------------
__global__ __launch_bounds__(256) void lmcl_norm_x(
    const float* __restrict__ x, _Float16* __restrict__ xn,
    float* __restrict__ denom, int D)
{
    const int row = blockIdx.x;
    const int t = threadIdx.x;
    const float4 v = ((const float4*)(x + (size_t)row * D))[t];
    float ss = v.x*v.x + v.y*v.y + v.z*v.z + v.w*v.w;
    #pragma unroll
    for (int off = 16; off; off >>= 1) ss += __shfl_xor(ss, off, 32);
    __shared__ float r8[8];
    if ((t & 31) == 0) r8[t >> 5] = ss;
    __syncthreads();
    float tot = 0.f;
    #pragma unroll
    for (int i = 0; i < 8; ++i) tot += r8[i];
    const float inv = rsqrtf(tot);
    v4h h;
    h[0] = (_Float16)(v.x * inv); h[1] = (_Float16)(v.y * inv);
    h[2] = (_Float16)(v.z * inv); h[3] = (_Float16)(v.w * inv);
    *(v4h*)(xn + (size_t)row * D + t * 4) = h;
    if (t == 0) denom[row] = 0.f;
}

// ---------------------------------------------------------------------------
// Kernel 2: fused normalize-W -> f16 LDS stage -> WMMA GEMM -> exp-sum epilogue.
// grid = ceil(C/BC) blocks, 128 threads (4 waves). Waves split the N dim.
// Each wave register-blocks 32 rows x 64 cols: every B fragment loaded from
// LDS feeds two WMMAs (ds:wmma ratio 1:1 instead of 2:1).
// ---------------------------------------------------------------------------
__global__ __launch_bounds__(128) void lmcl_gemm(
    const float* __restrict__ W, const float* __restrict__ bvec,
    const _Float16* __restrict__ xn, const int* __restrict__ labels,
    float* __restrict__ denom, float* __restrict__ ly,
    int N, int C, int D)
{
    __shared__ _Float16 Ws[BC * (1024 + WPAD)];   // 132 KB, f16 normalized W tile
    const int t    = threadIdx.x;
    const int lane = t & 31;
    const int wid  = t >> 5;
    const int c0   = blockIdx.x * BC;
    const int LSTR = D + WPAD;                    // 1032 halves per LDS row

    // ---- Stage: normalize BC rows of W into LDS as f16 (wave wid: rows 16w..16w+15)
    for (int rr = 0; rr < 16; ++rr) {
        const int lr = wid * 16 + rr;
        const int c  = c0 + lr;
        float4 v[8];
        if (c < C) {
            const float4* wr = (const float4*)(W + (size_t)c * D);
            #pragma unroll
            for (int q = 0; q < 8; ++q) v[q] = wr[q * 32 + lane];
        } else {
            #pragma unroll
            for (int q = 0; q < 8; ++q) v[q] = make_float4(0.f, 0.f, 0.f, 0.f);
        }
        float ss = 0.f;
        #pragma unroll
        for (int q = 0; q < 8; ++q)
            ss += v[q].x*v[q].x + v[q].y*v[q].y + v[q].z*v[q].z + v[q].w*v[q].w;
        #pragma unroll
        for (int off = 16; off; off >>= 1) ss += __shfl_xor(ss, off, 32);
        const float inv = rsqrtf(ss + 1e-24f);
        #pragma unroll
        for (int q = 0; q < 8; ++q) {
            v4h h;
            h[0] = (_Float16)(v[q].x * inv); h[1] = (_Float16)(v[q].y * inv);
            h[2] = (_Float16)(v[q].z * inv); h[3] = (_Float16)(v[q].w * inv);
            *(v4h*)(Ws + lr * LSTR + (q * 32 + lane) * 4) = h;
        }
    }
    __syncthreads();

    // Per-lane bias terms for this lane's 4 columns: logits = S*cos + (S+1)*b
    float sb[4];
    #pragma unroll
    for (int ct = 0; ct < 4; ++ct) {
        const int c = c0 + ct * 16 + (lane & 15);
        sb[ct] = (c < C) ? (S_SCALE + 1.0f) * bvec[c] : -1e30f;
    }

    const int KS  = D >> 5;               // k-steps of 32
    const int hiA = (lane >> 4) * 8;      // A frag: lanes 16-31 start at K+8
    const int hiB = (lane >> 4) * 8;      // B frag: lanes 16-31 start at K+8
    const int rA  = lane & 15;

    for (int m0 = wid * 32; m0 < N; m0 += 128) {   // 4 waves x 32 rows = 128/iter
        v8f acc[2][4] = {};
        const _Float16* arow0 = xn + (size_t)(m0 + rA) * D;
        const _Float16* arow1 = arow0 + (size_t)16 * D;
        for (int ks = 0; ks < KS; ++ks) {
            Frag16 fa0, fa1;
            const float4* ap0 = (const float4*)(arow0 + ks * 32 + hiA);
            const float4* ap1 = (const float4*)(arow1 + ks * 32 + hiA);
            fa0.f[0] = ap0[0]; fa0.f[1] = ap0[2];   // K = base+0..7, base+16..23
            fa1.f[0] = ap1[0]; fa1.f[1] = ap1[2];
            #pragma unroll
            for (int ct = 0; ct < 4; ++ct) {
                Frag16 fb;
                const float4* bp =
                    (const float4*)(Ws + (ct * 16 + rA) * LSTR + ks * 32 + hiB);
                fb.f[0] = bp[0];  // K = base + 0..7
                fb.f[1] = bp[2];  // K = base + 16..23
                acc[0][ct] = __builtin_amdgcn_wmma_f32_16x16x32_f16(
                    false, fa0.h, false, fb.h, (short)0, acc[0][ct], false, false);
                acc[1][ct] = __builtin_amdgcn_wmma_f32_16x16x32_f16(
                    false, fa1.h, false, fb.h, (short)0, acc[1][ct], false, false);
            }
        }
        // ---- Epilogue: logits, label capture, exp row-sums, atomic accumulate
        const int lab = labels[m0 + lane];   // labels for rows m0..m0+31
        #pragma unroll
        for (int nt = 0; nt < 2; ++nt) {
            #pragma unroll
            for (int r = 0; r < 8; ++r) {
                const int roff = nt * 16 + r + ((lane >> 4) << 3);
                const int mrow = m0 + roff;
                const int rlab = __shfl(lab, roff, 32);
                float s = 0.f;
                #pragma unroll
                for (int ct = 0; ct < 4; ++ct) {
                    const float lv = S_SCALE * acc[nt][ct][r] + sb[ct];
                    const int c = c0 + ct * 16 + (lane & 15);
                    if (c == rlab) ly[mrow] = lv;   // unique writer per row
                    s += __expf(lv);
                }
                s += __shfl_xor(s, 1, 32);
                s += __shfl_xor(s, 2, 32);
                s += __shfl_xor(s, 4, 32);
                s += __shfl_xor(s, 8, 32);
                if ((lane & 15) == 0) atomicAdd(denom + mrow, s);
            }
        }
    }
}

// ---------------------------------------------------------------------------
// Kernel 3: per-row loss and mean reduction. 1 block, 256 threads.
// ---------------------------------------------------------------------------
__global__ __launch_bounds__(256) void lmcl_final(
    const float* __restrict__ denom, const float* __restrict__ ly,
    float* __restrict__ out, int N)
{
    __shared__ float red[256];
    const float SM = S_SCALE * M_MARGIN;
    float s = 0.f;
    for (int i = threadIdx.x; i < N; i += 256) {
        const float l_y = ly[i];
        const float num = l_y - SM;                               // numerator
        const float den = denom[i] - __expf(l_y) + __expf(num);   // denominator
        s += num - __logf(den);
    }
    red[threadIdx.x] = s;
    __syncthreads();
    for (int off = 128; off; off >>= 1) {
        if (threadIdx.x < off) red[threadIdx.x] += red[threadIdx.x + off];
        __syncthreads();
    }
    if (threadIdx.x == 0) out[0] = -red[0] / (float)N;
}

// ---------------------------------------------------------------------------
extern "C" void kernel_launch(void* const* d_in, const int* in_sizes, int n_in,
                              void* d_out, int out_size, void* d_ws, size_t ws_size,
                              hipStream_t stream)
{
    const float* x      = (const float*)d_in[0];
    const int*   labels = (const int*)  d_in[1];
    const float* W      = (const float*)d_in[2];
    const float* b      = (const float*)d_in[3];
    const int N = in_sizes[1];
    const int D = in_sizes[0] / N;
    const int C = in_sizes[3];
    float* out = (float*)d_out;

    char* ws = (char*)d_ws;
    _Float16* xn = (_Float16*)ws;                       // N*D f16 (4 MB)
    float* denom = (float*)(ws + (size_t)N * D * sizeof(_Float16));
    float* ly    = denom + N;

    lmcl_norm_x<<<N, 256, 0, stream>>>(x, xn, denom, D);
    const int blocks = (C + BC - 1) / BC;
    lmcl_gemm<<<blocks, 128, 0, stream>>>(W, b, xn, labels, denom, ly, N, C, D);
    lmcl_final<<<1, 256, 0, stream>>>(denom, ly, out, N);
}
